// sqgkt_6579889897941
// MI455X (gfx1250) — matrix operand order repaired and checked by reading, not compiled, not run on hardware
//
#include <hip/hip_runtime.h>

#define B_  32
#define T_  100
#define D_  100
#define QN_ 5000
#define RK_ 10
#define NEGV -1e30f
#define LDP 112          // padded row stride for GEMM outputs (7*16)

typedef __attribute__((ext_vector_type(16))) _Float16 v16h;
typedef __attribute__((ext_vector_type(2)))  _Float16 v2h;
typedef __attribute__((ext_vector_type(8)))  float    v8f;

// ---------------- weight prep: f32 row-major -> f16 WMMA B-fragment tiles -------------
// Fragment (kt,nt) = 512 halfs. lane L, element e:  K = kt*32 + e + 16*(L>>4),
// N = nt*16 + (L&15).  Row-stacked sources A (Ka rows) then B (Kb rows), stride N.
__global__ void prep_weights(const float* __restrict__ srcA, int Ka,
                             const float* __restrict__ srcB, int Kb,
                             int N, _Float16* __restrict__ dst, int KT, int NT) {
  int idx = blockIdx.x * blockDim.x + threadIdx.x;
  int total = KT * NT * 512;
  if (idx >= total) return;
  int f    = idx >> 9;
  int r    = idx & 511;
  int lane = r >> 4;
  int e    = r & 15;
  int kt   = f % KT;
  int nt   = f / KT;
  int k = kt * 32 + e + 16 * (lane >> 4);
  int n = nt * 16 + (lane & 15);
  float v = 0.0f;
  if (n < N) {
    if (k < Ka)            v = srcA[(size_t)k * N + n];
    else if (k - Ka < Kb)  v = srcB[(size_t)(k - Ka) * N + n];
  }
  dst[idx] = (_Float16)v;
}

// bias pad/combine:  dst[0..NP) = (i<N) ? a[i] (+ b[i]) : 0
__global__ void prep_bias(const float* __restrict__ a, const float* __restrict__ b,
                          int N, int NP, float* __restrict__ dst) {
  int i = blockIdx.x * blockDim.x + threadIdx.x;
  if (i >= NP) return;
  float v = 0.f;
  if (i < N) { v = a[i]; if (b) v += b[i]; }
  dst[i] = v;
}

// A-fragment slot for logical element (m,k) with k even; (m,k+1) lands in slot+1.
// A layout used here (contiguous k, same as B prep): lane = 16*h + (m&15), kl = e + 16h.
__device__ __forceinline__ int afrag_idx(int m, int k, int KT) {
  int mt = m >> 4, mrow = m & 15;
  int kt = k >> 5, kl = k & 31;
  int h  = kl >> 4;
  int e  = kl & 15;
  return (((mt * KT + kt) << 9) + ((h * 16 + mrow) << 4) + e);
}

__device__ __forceinline__ void store_pair(_Float16* sA, int slot, float v0, float v1) {
  v2h p;
  p[0] = (_Float16)v0;
  p[1] = (_Float16)v1;
  *(v2h*)(sA + slot) = p;   // even slot -> 4B aligned -> ds_store_b32
}

// ---------------- WMMA GEMM on pre-staged fragments -----------------------------------
// sA: LDS f16 A-fragments (MT*KT frags). W: global f16 B-fragments.
// biasPad: padded to NT*16. Y: padded LDS, 16-row aligned, stride ldy.
// Stores fully unguarded; rows >= M hold deterministic garbage nobody reads.
template<int KT, int NT>
__device__ __forceinline__ void wmma_gemm_fr(
    const _Float16* __restrict__ sA, int M,
    const _Float16* __restrict__ W,
    const float* __restrict__ biasPad,
    float* __restrict__ Y, int ldy, bool relu, int tid) {
  const int lane = tid & 31;
  const int wave = tid >> 5;
  const int hh   = lane >> 4;
  const int mrow = lane & 15;
  const int MT   = (M + 15) >> 4;
  for (int mt = 0; mt < MT; ++mt) {
    v16h a[KT];
#pragma unroll
    for (int kt = 0; kt < KT; ++kt)
      a[kt] = *(const v16h*)(sA + (((mt * KT + kt) << 9) + (lane << 4)));
    for (int nt = wave; nt < NT; nt += 4) {
      v8f acc = {0.f, 0.f, 0.f, 0.f, 0.f, 0.f, 0.f, 0.f};
#pragma unroll
      for (int kt = 0; kt < KT; ++kt) {
        const v16h bf = *(const v16h*)(W + (((size_t)(nt * KT + kt)) << 9) + (lane << 4));
        acc = __builtin_amdgcn_wmma_f32_16x16x32_f16(false, a[kt], false, bf,
                                                     (short)0, acc, false, false);
      }
      const int n = nt * 16 + mrow;
      const float bv = biasPad[n];
#pragma unroll
      for (int r = 0; r < 8; ++r) {
        float v = acc[r] + bv;
        if (relu) v = fmaxf(v, 0.0f);
        Y[(mt * 16 + r + 8 * hh) * ldy + n] = v;
      }
    }
  }
  __syncthreads();
}

// ---------------- one 3-hop aggregation side ----------------
__device__ __forceinline__ void agg_side(int root,
                         const int* __restrict__ tabA, const int* __restrict__ tabB,
                         const float* __restrict__ embA, const float* __restrict__ embB,
                         const float* __restrict__ uq,   // null => unweighted
                         float wc, float wp, float wn,
                         const _Float16* wAgg0, const _Float16* wAgg1,
                         const _Float16* wAgg2, const _Float16* wAggL,
                         const float* __restrict__ bAgg,   // 3*LDP padded
                         const float* __restrict__ bAggL,  // LDP padded
                         _Float16* sA, float* sL2, float* sL1, float* sL0,
                         int* sid1, int* sid2, int* sid3,
                         float* sfw1, float* sfw2, float* sfw3,
                         float* outAgg, int tid) {
  for (int i = tid; i < 8; i += 128)   sid1[i] = tabA[root * 8 + i];
  __syncthreads();
  for (int i = tid; i < 64; i += 128)  sid2[i] = tabB[sid1[i >> 3] * 8 + (i & 7)];
  __syncthreads();
  for (int i = tid; i < 512; i += 128) sid3[i] = tabA[sid2[i >> 3] * 8 + (i & 7)];
  __syncthreads();
  // prefetch level-3 rows (consumed by pass0/j=2 builder) -> global_prefetch_b8
  for (int i = tid; i < 512; i += 128)
    __builtin_prefetch(embB + (size_t)sid3[i] * D_, 0, 0);
  for (int i = tid; i < D_; i += 128) sL0[i] = embA[(size_t)root * D_ + i];
  for (int i = tid; i < 8 * 50; i += 128) {
    int r = i / 50, c = (i % 50) * 2;
    *(float2*)&sL1[r * LDP + c] = *(const float2*)&embB[(size_t)sid1[r] * D_ + c];
  }
  for (int i = tid; i < 64 * 50; i += 128) {
    int r = i / 50, c = (i % 50) * 2;
    *(float2*)&sL2[r * LDP + c] = *(const float2*)&embA[(size_t)sid2[r] * D_ + c];
  }
  if (uq) {
    for (int i = tid; i < 8 + 64 + 512; i += 128) {
      int uid, qid; float* dst;
      if (i < 8)        { uid = root;            qid = sid1[i];            dst = &sfw1[i]; }
      else if (i < 72)  { int j = i - 8;  uid = sid2[j];  qid = sid1[j >> 3]; dst = &sfw2[j]; }
      else              { int r = i - 72; uid = sid2[r >> 3]; qid = sid3[r];  dst = &sfw3[r]; }
      const float* w = uq + ((size_t)uid * QN_ + qid) * 3;
      *dst = wc * w[0] + wp * w[1] + wn * w[2];
    }
  }
  __syncthreads();
  const _Float16* aggWs[3] = {wAgg0, wAgg1, wAgg2};
  for (int pass = 0; pass < 3; ++pass) {
    const int maxj = 3 - pass;
    for (int j = 0; j < maxj; ++j) {
      const int M  = (j == 0) ? 1 : (j == 1) ? 8 : 64;
      const int Mp = (M + 15) & ~15;
      const float* self = (j == 0) ? sL0 : (j == 1) ? sL1 : sL2;
      const float* fw = uq ? ((j == 0) ? sfw1 : (j == 1) ? sfw2 : sfw3) : nullptr;
      if (j < 2) {
        const float* nxt = (j == 0) ? sL1 : sL2;
        for (int idx = tid; idx < Mp * 64; idx += 128) {
          int m = idx >> 6, k0 = (idx & 63) << 1;
          float v0 = 0.f, v1 = 0.f;
          if (m < M && k0 < D_) {
            float s0 = 0.f, s1 = 0.f;
            for (int k2 = 0; k2 < 8; ++k2) {
              const float* row = nxt + (m * 8 + k2) * LDP + k0;
              float x0 = row[0], x1 = row[1];
              if (fw) { float f = fw[m * 8 + k2]; x0 *= f; x1 *= f; }
              s0 += x0; s1 += x1;
            }
            v0 = s0 * 0.125f + self[m * LDP + k0];
            v1 = s1 * 0.125f + self[m * LDP + k0 + 1];
          }
          store_pair(sA, afrag_idx(m, k0, 4), v0, v1);
        }
      } else {
        for (int idx = tid; idx < Mp * 64; idx += 128) {
          int m = idx >> 6, k0 = (idx & 63) << 1;
          float v0 = 0.f, v1 = 0.f;
          if (m < M && k0 < D_) {
            float s0 = 0.f, s1 = 0.f;
            for (int k2 = 0; k2 < 8; ++k2) {
              const float* row = embB + (size_t)sid3[m * 8 + k2] * D_ + k0;
              float x0 = row[0], x1 = row[1];
              if (fw) { float f = fw[m * 8 + k2]; x0 *= f; x1 *= f; }
              s0 += x0; s1 += x1;
            }
            v0 = s0 * 0.125f + self[m * LDP + k0];
            v1 = s1 * 0.125f + self[m * LDP + k0 + 1];
          }
          store_pair(sA, afrag_idx(m, k0, 4), v0, v1);
        }
      }
      __syncthreads();
      float* Y = (j == 0) ? sL0 : (j == 1) ? sL1 : sL2;
      wmma_gemm_fr<4, 7>(sA, M, aggWs[j], bAgg + j * LDP, Y, LDP, true, tid);
    }
  }
  // final layer: M=1, K=D
  for (int idx = tid; idx < 16 * 64; idx += 128) {
    int m = idx >> 6, k0 = (idx & 63) << 1;
    float v0 = 0.f, v1 = 0.f;
    if (m == 0 && k0 < D_) { v0 = sL0[k0]; v1 = sL0[k0 + 1]; }
    store_pair(sA, afrag_idx(m, k0, 4), v0, v1);
  }
  __syncthreads();
  wmma_gemm_fr<4, 7>(sA, 1, wAggL, bAggL, outAgg, LDP, true, tid);
}

struct Params {
  const int *user, *question, *response;
  const int *q_nb, *s_nb, *u_nb, *q_nb2, *skill_tab;
  const float *uq_table;
  const float *emb_q, *emb_q2, *emb_s, *emb_u, *emb_r;
  const float *w1, *w2, *wc, *wp, *wn;
  const float *attn_W, *attn_b;
  const _Float16 *wAgg0, *wAgg1, *wAgg2, *wAggL, *wFus, *wLstm, *wQry, *wKey;
  const float *bAgg, *bAggL, *bFus, *bLstm, *bQry, *bKey;  // padded biases in ws
  float *hist;   // B*T*D scratch
  float *out;    // B*T
};

__global__ __launch_bounds__(128) void sqgkt_main(Params P) {
  const int b   = blockIdx.x;
  const int tid = threadIdx.x;

  __shared__ alignas(64) _Float16 sA[16 * 512];   // A-fragment staging (16 KB)
  __shared__ float sL2[64 * LDP];
  __shared__ float sL1[16 * LDP];
  __shared__ float sL0[16 * LDP];
  __shared__ float sQagg[16 * LDP], sUagg[16 * LDP], sE[16 * LDP];
  __shared__ float sKK[16 * LDP], sQQ[16 * LDP];
  __shared__ float sGates[16 * 400];
  __shared__ float sH[D_], sC[D_];
  __shared__ int   sid1[8], sid2[64], sid3[512];
  __shared__ float sfw1[8], sfw2[64], sfw3[512];
  __shared__ float sQS[5 * D_];
  __shared__ float sHist[11 * D_];
  __shared__ float sScore[T_];
  __shared__ float sTopV[RK_];
  __shared__ int   sTopI[RK_];
  __shared__ float sKa[5], sQa[11], sLg[55], sInner[55];
  __shared__ float sRedV[128];
  __shared__ int   sRedI[128];
  __shared__ int   sQt, sUt, sRt, sQn;

  float* myhist = P.hist + (size_t)b * T_ * D_;
  for (int i = tid; i < T_ * D_; i += 128) myhist[i] = 0.f;
  for (int i = tid; i < D_; i += 128) { sH[i] = 0.f; sC[i] = 0.f; }
  if (tid == 0) P.out[b * T_] = 0.f;
  __threadfence_block();
  __syncthreads();

  const float w1 = P.w1[0], w2 = P.w2[0];
  const float wc = P.wc[0], wp = P.wp[0], wn = P.wn[0];

  for (int t = 0; t < T_ - 1; ++t) {
    if (tid == 0) {
      sQt = P.question[b * T_ + t];
      sUt = P.user[b * T_ + t];
      sRt = P.response[b * T_ + t];
      sQn = P.question[b * T_ + t + 1];
    }
    __syncthreads();

    // ---- question-side aggregate (unweighted) ----
    agg_side(sQt, P.q_nb, P.s_nb, P.emb_q, P.emb_s, nullptr, wc, wp, wn,
             P.wAgg0, P.wAgg1, P.wAgg2, P.wAggL, P.bAgg, P.bAggL,
             sA, sL2, sL1, sL0, sid1, sid2, sid3, sfw1, sfw2, sfw3, sQagg, tid);
    // ---- user-side aggregate (uq-weighted) ----
    agg_side(sUt, P.u_nb, P.q_nb2, P.emb_u, P.emb_q2, P.uq_table, wc, wp, wn,
             P.wAgg0, P.wAgg1, P.wAgg2, P.wAggL, P.bAgg, P.bAggL,
             sA, sL2, sL1, sL0, sid1, sid2, sid3, sfw1, sfw2, sfw3, sUagg, tid);

    // ---- fusion:  e = relu([w1*qagg + w2*uagg, emb_r[r]] @ Wf + bf);  K=200,KT=7 ----
    for (int idx = tid; idx < 16 * 112; idx += 128) {
      int m = idx / 112, k0 = (idx % 112) * 2;
      float v0 = 0.f, v1 = 0.f;
      if (m == 0) {
        if (k0 < D_) {
          v0 = w1 * sQagg[k0] + w2 * sUagg[k0];
          v1 = w1 * sQagg[k0 + 1] + w2 * sUagg[k0 + 1];
        } else if (k0 < 2 * D_) {
          v0 = P.emb_r[sRt * D_ + (k0 - D_)];
          v1 = P.emb_r[sRt * D_ + (k0 + 1 - D_)];
        }
      }
      store_pair(sA, afrag_idx(m, k0, 7), v0, v1);
    }
    __syncthreads();
    wmma_gemm_fr<7, 7>(sA, 1, P.wFus, P.bFus, sE, LDP, true, tid);

    // ---- LSTM: gates = [e, h] @ [Wih;Whh] + (bih + bhh);  K=200,KT=7, N=400,NT=25 ----
    for (int idx = tid; idx < 16 * 112; idx += 128) {
      int m = idx / 112, k0 = (idx % 112) * 2;
      float v0 = 0.f, v1 = 0.f;
      if (m == 0) {
        if (k0 < D_)          { v0 = sE[k0];      v1 = sE[k0 + 1]; }
        else if (k0 < 2 * D_) { v0 = sH[k0 - D_]; v1 = sH[k0 + 1 - D_]; }
      }
      store_pair(sA, afrag_idx(m, k0, 7), v0, v1);
    }
    __syncthreads();
    wmma_gemm_fr<7, 25>(sA, 1, P.wLstm, P.bLstm, sGates, 400, false, tid);
    for (int d = tid; d < D_; d += 128) {
      float ig = sGates[d], fg = sGates[D_ + d], gg = sGates[2 * D_ + d], og = sGates[3 * D_ + d];
      float si = 1.f / (1.f + __expf(-ig));
      float sf = 1.f / (1.f + __expf(-fg));
      float so = 1.f / (1.f + __expf(-og));
      float cn = sf * sC[d] + si * tanhf(gg);
      float hn = so * tanhf(cn);
      sC[d] = cn; sH[d] = hn;
      myhist[t * D_ + d] = hn;
    }
    __threadfence_block();
    __syncthreads();

    // ---- prediction head ----
    for (int i = tid; i < 5 * D_; i += 128) {
      int r = i / D_, d = i % D_;
      float v;
      if (r == 0) v = P.emb_q[(size_t)sQn * D_ + d];
      else {
        int sk = P.skill_tab[sQn * 4 + (r - 1)];
        v = P.emb_s[(size_t)sk * D_ + d];
      }
      sQS[i] = v;
    }
    __syncthreads();
    for (int tt = tid; tt < T_; tt += 128) {
      const float* row = P.emb_q + (size_t)P.question[b * T_ + tt] * D_;
      float s = 0.f;
      for (int d = 0; d < D_; ++d) s += row[d] * sQS[d];
      sScore[tt] = (tt < t) ? s : NEGV;
    }
    __syncthreads();
    // top-k (10 rounds of argmax; stable: smaller index wins ties)
    for (int r = 0; r < RK_; ++r) {
      float bv = -3.0e38f; int bi = 0;
      for (int tt = tid; tt < T_; tt += 128) {
        if (sScore[tt] > bv) { bv = sScore[tt]; bi = tt; }
      }
      sRedV[tid] = bv; sRedI[tid] = bi;
      __syncthreads();
      for (int s2 = 64; s2 > 0; s2 >>= 1) {
        if (tid < s2) {
          float ov = sRedV[tid + s2]; int oi = sRedI[tid + s2];
          if (ov > sRedV[tid] || (ov == sRedV[tid] && oi < sRedI[tid])) {
            sRedV[tid] = ov; sRedI[tid] = oi;
          }
        }
        __syncthreads();
      }
      if (tid == 0) { sTopV[r] = sRedV[0]; sTopI[r] = sRedI[0]; sScore[sRedI[0]] = -3.0e38f; }
      __syncthreads();
    }
    for (int i = tid; i < 11 * D_; i += 128) {
      int r = i / D_, d = i % D_;
      sHist[i] = (r == 0) ? sH[d] : myhist[sTopI[r - 1] * D_ + d];
    }
    __syncthreads();
    // key: 5xD @ key_W ; query: 11xD @ query_W
    for (int idx = tid; idx < 16 * 64; idx += 128) {
      int m = idx >> 6, k0 = (idx & 63) << 1;
      float v0 = 0.f, v1 = 0.f;
      if (m < 5 && k0 < D_) { v0 = sQS[m * D_ + k0]; v1 = sQS[m * D_ + k0 + 1]; }
      store_pair(sA, afrag_idx(m, k0, 4), v0, v1);
    }
    __syncthreads();
    wmma_gemm_fr<4, 7>(sA, 5, P.wKey, P.bKey, sKK, LDP, false, tid);
    for (int idx = tid; idx < 16 * 64; idx += 128) {
      int m = idx >> 6, k0 = (idx & 63) << 1;
      float v0 = 0.f, v1 = 0.f;
      if (m < 11 && k0 < D_) { v0 = sHist[m * D_ + k0]; v1 = sHist[m * D_ + k0 + 1]; }
      store_pair(sA, afrag_idx(m, k0, 4), v0, v1);
    }
    __syncthreads();
    wmma_gemm_fr<4, 7>(sA, 11, P.wQry, P.bQry, sQQ, LDP, false, tid);
    if (tid < 16) {
      if (tid < 5) {
        float s = 0.f;
        for (int d = 0; d < D_; ++d) s += sKK[tid * LDP + d] * P.attn_W[d];
        sKa[tid] = s;
      } else {
        int j = tid - 5;
        float s = 0.f;
        for (int d = 0; d < D_; ++d) s += sQQ[j * LDP + d] * P.attn_W[D_ + d];
        sQa[j] = s;
      }
    }
    __syncthreads();
    for (int i = tid; i < 55; i += 128) {
      int qi = i / 11, hj = i % 11;
      bool valid = (hj == 0) || (sTopV[hj - 1] > NEGV * 0.5f);
      sLg[i] = valid ? (sKa[qi] + sQa[hj] + P.attn_b[0]) : NEGV;
      float s = 0.f;
      for (int d = 0; d < D_; ++d) s += sQS[qi * D_ + d] * sHist[hj * D_ + d];
      sInner[i] = s;
    }
    __syncthreads();
    if (tid == 0) {
      float mx = -3.0e38f;
      for (int i = 0; i < 55; ++i) mx = fmaxf(mx, sLg[i]);
      float den = 0.f, num = 0.f;
      for (int i = 0; i < 55; ++i) {
        float e = __expf(sLg[i] - mx);
        den += e; num += e * sInner[i];
      }
      float y = num / den;
      P.out[b * T_ + t + 1] = 1.f / (1.f + __expf(-y));
    }
    __syncthreads();
  }
}

// ---------------------------------------------------------------------------
extern "C" void kernel_launch(void* const* d_in, const int* in_sizes, int n_in,
                              void* d_out, int out_size, void* d_ws, size_t ws_size,
                              hipStream_t stream) {
  (void)in_sizes; (void)n_in; (void)out_size; (void)ws_size;
  const int   *user    = (const int*)d_in[0];
  const int   *question= (const int*)d_in[1];
  const int   *response= (const int*)d_in[2];
  const int   *q_nb    = (const int*)d_in[4];
  const int   *s_nb    = (const int*)d_in[5];
  const int   *u_nb    = (const int*)d_in[6];
  const int   *q_nb2   = (const int*)d_in[7];
  const int   *skill   = (const int*)d_in[8];
  const float *uq      = (const float*)d_in[9];
  const float *emb_q   = (const float*)d_in[10];
  const float *emb_q2  = (const float*)d_in[11];
  const float *emb_s   = (const float*)d_in[12];
  const float *emb_u   = (const float*)d_in[13];
  const float *emb_r   = (const float*)d_in[14];
  const float *w1      = (const float*)d_in[15];
  const float *w2      = (const float*)d_in[16];
  const float *wc      = (const float*)d_in[17];
  const float *wp      = (const float*)d_in[18];
  const float *wn      = (const float*)d_in[19];
  const float *lstm_Wih= (const float*)d_in[20];
  const float *lstm_Whh= (const float*)d_in[21];
  const float *lstm_bih= (const float*)d_in[22];
  const float *lstm_bhh= (const float*)d_in[23];
  const float *fusion_W= (const float*)d_in[24];
  const float *fusion_b= (const float*)d_in[25];
  const float *agg_W   = (const float*)d_in[26];
  const float *agg_b   = (const float*)d_in[27];
  const float *agg_lW  = (const float*)d_in[28];
  const float *agg_lb  = (const float*)d_in[29];
  const float *query_W = (const float*)d_in[30];
  const float *query_b = (const float*)d_in[31];
  const float *key_W   = (const float*)d_in[32];
  const float *key_b   = (const float*)d_in[33];
  const float *attn_W  = (const float*)d_in[34];
  const float *attn_b  = (const float*)d_in[35];

  // d_ws carve-up: tiled f16 weights, then padded f32 biases, then f32 state_hist
  _Float16* wbase = (_Float16*)d_ws;
  _Float16* wAgg0 = wbase + 0;        // 4*7*512  = 14336
  _Float16* wAgg1 = wbase + 14336;
  _Float16* wAgg2 = wbase + 28672;
  _Float16* wAggL = wbase + 43008;
  _Float16* wFus  = wbase + 57344;    // 7*7*512  = 25088
  _Float16* wLstm = wbase + 82432;    // 7*25*512 = 89600
  _Float16* wQry  = wbase + 172032;
  _Float16* wKey  = wbase + 186368;
  float*    fbase = (float*)(wbase + 200704);
  float* bAgg  = fbase;               // 3*LDP
  float* bAggL = fbase + 3 * LDP;
  float* bFus  = fbase + 4 * LDP;
  float* bQry  = fbase + 5 * LDP;
  float* bKey  = fbase + 6 * LDP;
  float* bLstm = fbase + 7 * LDP;     // 400
  float* hist  = fbase + 7 * LDP + 400;   // B*T*D floats

  auto prep = [&](const float* A, int Ka, const float* Bsrc, int Kb, int N,
                  _Float16* dst, int KT, int NT) {
    int total = KT * NT * 512;
    prep_weights<<<(total + 255) / 256, 256, 0, stream>>>(A, Ka, Bsrc, Kb, N, dst, KT, NT);
  };
  prep(agg_W + 0 * D_ * D_, D_, nullptr, 0, D_,     wAgg0, 4, 7);
  prep(agg_W + 1 * D_ * D_, D_, nullptr, 0, D_,     wAgg1, 4, 7);
  prep(agg_W + 2 * D_ * D_, D_, nullptr, 0, D_,     wAgg2, 4, 7);
  prep(agg_lW,              D_, nullptr, 0, D_,     wAggL, 4, 7);
  prep(fusion_W,        2 * D_, nullptr, 0, D_,     wFus,  7, 7);
  prep(lstm_Wih,            D_, lstm_Whh, D_, 4*D_, wLstm, 7, 25);
  prep(query_W,             D_, nullptr, 0, D_,     wQry,  4, 7);
  prep(key_W,               D_, nullptr, 0, D_,     wKey,  4, 7);

  prep_bias<<<1, LDP, 0, stream>>>(agg_b + 0 * D_, nullptr, D_, LDP, bAgg + 0 * LDP);
  prep_bias<<<1, LDP, 0, stream>>>(agg_b + 1 * D_, nullptr, D_, LDP, bAgg + 1 * LDP);
  prep_bias<<<1, LDP, 0, stream>>>(agg_b + 2 * D_, nullptr, D_, LDP, bAgg + 2 * LDP);
  prep_bias<<<1, LDP, 0, stream>>>(agg_lb,         nullptr, D_, LDP, bAggL);
  prep_bias<<<1, LDP, 0, stream>>>(fusion_b,       nullptr, D_, LDP, bFus);
  prep_bias<<<1, LDP, 0, stream>>>(query_b,        nullptr, D_, LDP, bQry);
  prep_bias<<<1, LDP, 0, stream>>>(key_b,          nullptr, D_, LDP, bKey);
  prep_bias<<<2, 256, 0, stream>>>(lstm_bih,  lstm_bhh, 4 * D_, 4 * D_, bLstm);

  Params P;
  P.user = user; P.question = question; P.response = response;
  P.q_nb = q_nb; P.s_nb = s_nb; P.u_nb = u_nb; P.q_nb2 = q_nb2; P.skill_tab = skill;
  P.uq_table = uq;
  P.emb_q = emb_q; P.emb_q2 = emb_q2; P.emb_s = emb_s; P.emb_u = emb_u; P.emb_r = emb_r;
  P.w1 = w1; P.w2 = w2; P.wc = wc; P.wp = wp; P.wn = wn;
  P.attn_W = attn_W; P.attn_b = attn_b;
  P.wAgg0 = wAgg0; P.wAgg1 = wAgg1; P.wAgg2 = wAgg2; P.wAggL = wAggL;
  P.wFus = wFus; P.wLstm = wLstm; P.wQry = wQry; P.wKey = wKey;
  P.bAgg = bAgg; P.bAggL = bAggL; P.bFus = bFus; P.bLstm = bLstm;
  P.bQry = bQry; P.bKey = bKey;
  P.hist = hist; P.out = (float*)d_out;

  sqgkt_main<<<B_, 128, 0, stream>>>(P);
}
